// SimpleSSM_47983374630968
// MI455X (gfx1250) — matrix operand order repaired and checked by reading, not compiled, not run on hardware
//
#include <hip/hip_runtime.h>
#include <hip/hip_bf16.h>

// ---------------------------------------------------------------------------
// S4-style SSM: 128-tap causal FIR conv (with the reference's 127-sample
// circular-wrap delay), cast as bf16 WMMA GEMMs with hi/lo error compensation.
// Data path: global f32 -> (convert exactly once) -> wave-private LDS ring
// (bf16 hi/lo, mirrored tail) -> ds_load Toeplitz gathers with immediate
// offsets -> v_wmma_f32_16x16x32_bf16 (f32 accumulate).
// ---------------------------------------------------------------------------

typedef __attribute__((ext_vector_type(16))) __bf16 v16bf;
typedef __attribute__((ext_vector_type(8)))  __bf16 v8bf;
typedef __attribute__((ext_vector_type(8)))  float  v8f;
typedef __attribute__((ext_vector_type(4)))  float  v4f;

#define T_LEN           1048576
#define KLEN_           128
#define N2_             32
#define BATCH_          32
#define NCHUNK          5        // ceil(143/32) K=32 chunks per 16x16 tile
#define TILES_PER_WAVE  16
#define WAVES_PER_BLOCK 8
#define SEGS_PER_BATCH  256      // (T/256 tiles) / TILES_PER_WAVE
#define RING_ROW        920      // 512 ring + 400 mirror, padded (1840 B)

// Workspace layout (d_ws):
//   [    0,   512)  float  h[128]
//   [  512,  5632)  __bf16 Bhi[5][512]   (B operands, WMMA lane layout)
//   [ 5632, 10752)  __bf16 Blo[5][512]

// ---------------------------------------------------------------------------
// Kernel 1: build h[k] = Re(sum_n C_n*B_n*exp(k*dt*w_n)), h[0]+=D, then emit
// the 5 Toeplitz B-matrices B_s[kk][n] = h[128-32s+n-kk] in hi/lo bf16,
// pre-swizzled into the wave32 WMMA B-operand layout:
//   lane l -> column n = l%16 ; K = (e&7) + 8*(l/16) + 16*(e/8), e = element 0..15.
// ---------------------------------------------------------------------------
__global__ __launch_bounds__(256) void ssm_build_h(
    const float* __restrict__ wr, const float* __restrict__ wi,
    const float* __restrict__ log_dt,
    const float* __restrict__ Cr, const float* __restrict__ Ci,
    const float* __restrict__ Br, const float* __restrict__ Bi,
    const float* __restrict__ Dp,
    float* __restrict__ h_out,
    __bf16* __restrict__ bhi, __bf16* __restrict__ blo)
{
    __shared__ float h_s[KLEN_];
    const int tid = threadIdx.x;
    const float dt = expf(log_dt[0]);

    if (tid < KLEN_) {
        const float kf = (float)tid;
        float acc = 0.f;
        for (int n = 0; n < N2_; ++n) {
            const float cr = Cr[n], ci = Ci[n], br = Br[n], bi = Bi[n];
            const float coef_r = cr * br - ci * bi;
            const float coef_i = cr * bi + ci * br;
            const float a = dt * wr[n] * kf;    // decaying real part
            const float b = dt * wi[n] * kf;    // phase
            acc += expf(a) * (coef_r * cosf(b) - coef_i * sinf(b));
        }
        if (tid == 0) acc += Dp[0];
        h_s[tid]  = acc;
        h_out[tid] = acc;
    }
    __syncthreads();

    for (int p = tid; p < NCHUNK * 512; p += 256) {
        const int s    = p >> 9;
        const int r    = p & 511;
        const int lane = r >> 4;
        const int e    = r & 15;
        const int n    = lane & 15;
        const int K    = (e & 7) + ((lane < 16) ? 0 : 8) + ((e < 8) ? 0 : 16);
        const int hidx = 128 - 32 * s + n - K;
        const float v  = (hidx >= 0 && hidx < KLEN_) ? h_s[hidx] : 0.f;
        const __bf16 hv = (__bf16)v;
        bhi[p] = hv;
        blo[p] = (__bf16)(v - (float)hv);
    }
}

// ---------------------------------------------------------------------------
// Stage 8 consecutive x values per lane into the bf16 hi/lo ring.
// Ring slot = key & 511; slots < 400 are mirrored at +512 so that gathers
// can address sbase+idx (<= 910) without any per-access wrap mask.
// ---------------------------------------------------------------------------
__device__ __forceinline__ void stage8(const float* __restrict__ xb, long key0,
                                       bool guard,
                                       __bf16* __restrict__ hiRow,
                                       __bf16* __restrict__ loRow)
{
    float f[8];
    if (!guard) {
        const v4f* p = (const v4f*)(xb + key0);      // 32B-aligned
        v4f a0 = p[0], a1 = p[1];
#pragma unroll
        for (int e = 0; e < 4; ++e) { f[e] = a0[e]; f[4 + e] = a1[e]; }
    } else {
#pragma unroll
        for (int e = 0; e < 8; ++e) {
            const long j = key0 + e;
            f[e] = (j >= 0 && j < T_LEN) ? xb[j] : 0.f;
        }
    }
    v8bf hv, lv;
#pragma unroll
    for (int e = 0; e < 8; ++e) {
        const __bf16 h = (__bf16)f[e];               // RNE hi
        hv[e] = h;
        lv[e] = (__bf16)(f[e] - (float)h);           // exact residual -> bf16
    }
    const int slot = (int)(key0 & 511);              // 8-aligned, no run split
    *(v8bf*)&hiRow[slot] = hv;
    *(v8bf*)&loRow[slot] = lv;
    if (slot < 400) {                                // mirror tail
        *(v8bf*)&hiRow[slot + 512] = hv;
        *(v8bf*)&loRow[slot + 512] = lv;
    }
}

// ---------------------------------------------------------------------------
// Kernel 2: streaming WMMA convolution with a per-wave LDS ring.
// Each wave owns 16 (or 17) consecutive 256-output tiles of one batch row.
// Per tile:
//   stage : 2x global_load_b128 (the 256 NEW floats only) -> hi/lo bf16
//           split -> 2-4x ds_store_b128 into the wave-private ring
//   gather: per chunk, 4x ds_load_b128 (A hi+lo, dense 16-bit A lane layout:
//           lanes 0-15 take K 0-7,16-23; lanes 16-31 take K 8-15,24-31),
//           all immediate offsets off one per-tile base
//   math  : 3 compensated v_wmma_f32_16x16x32_bf16 per chunk (f32 accum)
// Wave-private rings + in-order LDS => no barriers in the loop.
// ---------------------------------------------------------------------------
__global__ __launch_bounds__(256, 1) void ssm_conv_wmma(
    const float* __restrict__ x,
    const __bf16* __restrict__ bhi_g,
    const __bf16* __restrict__ blo_g,
    float* __restrict__ out)
{
    __shared__ __bf16 sHi[WAVES_PER_BLOCK][RING_ROW];
    __shared__ __bf16 sLo[WAVES_PER_BLOCK][RING_ROW];

    const int lane  = threadIdx.x & 31;
    const int wave  = threadIdx.x >> 5;
    const int wgid  = blockIdx.x * WAVES_PER_BLOCK + wave;
    const int batch = wgid >> 8;                 // 256 wave-segments per batch
    const int seg   = wgid & 255;
    const long xbase = (long)batch * T_LEN;
    const float* xb  = x + xbase;

    const int m    = lane & 15;                  // A row M, also D column N
    const int half = lane >> 4;

    __bf16* hiRow = sHi[wave];
    __bf16* loRow = sLo[wave];

    // B operands resident in VGPRs for the whole wave lifetime (L2 broadcast).
    v16bf Bh[NCHUNK], Bl[NCHUNK];
    {
        const v16bf* bh = (const v16bf*)bhi_g;
        const v16bf* bl = (const v16bf*)blo_g;
#pragma unroll
        for (int s = 0; s < NCHUNK; ++s) {
            Bh[s] = bh[s * 32 + lane];
            Bl[s] = bl[s * 32 + lane];
        }
    }

    const long t0f = (long)seg * (TILES_PER_WAVE * 256);

    // Prologue: stage [t0f-128, t0f+127] (256 keys, 8 per lane).
    stage8(xb, t0f - 128 + lane * 8, seg == 0, hiRow, loRow);

    // Last segment also computes the extra t0 == T tile (wrap region).
    const int ntiles = TILES_PER_WAVE + ((seg == SEGS_PER_BATCH - 1) ? 1 : 0);

    for (int it = 0; it < ntiles; ++it) {
        const long t0 = t0f + (long)it * 256;

        // Stage the 256 new keys [t0+128, t0+383].
        stage8(xb, t0 + 128 + lane * 8, (t0 + 384 > T_LEN), hiRow, loRow);

        // ---- gather + WMMA ----
        const int sbase = (int)((t0 - 128) & 511);   // uniform per tile
        const __bf16* gH = hiRow + sbase + 16 * m + 8 * half;
        const __bf16* gL = loRow + sbase + 16 * m + 8 * half;

        v8f acc = {0.f, 0.f, 0.f, 0.f, 0.f, 0.f, 0.f, 0.f};
#pragma unroll
        for (int s = 0; s < NCHUNK; ++s) {
            v8bf ah0 = *(const v8bf*)(gH + 32 * s);
            v8bf ah1 = *(const v8bf*)(gH + 32 * s + 16);
            v8bf al0 = *(const v8bf*)(gL + 32 * s);
            v8bf al1 = *(const v8bf*)(gL + 32 * s + 16);
            v16bf ah = __builtin_shufflevector(ah0, ah1,
                0, 1, 2, 3, 4, 5, 6, 7, 8, 9, 10, 11, 12, 13, 14, 15);
            v16bf al = __builtin_shufflevector(al0, al1,
                0, 1, 2, 3, 4, 5, 6, 7, 8, 9, 10, 11, 12, 13, 14, 15);

            // acc += x_hi*h_hi + x_lo*h_hi + x_hi*h_lo  (f32 accumulate)
            acc = __builtin_amdgcn_wmma_f32_16x16x32_bf16(
                false, ah, false, Bh[s], (short)0, acc, false, false);
            acc = __builtin_amdgcn_wmma_f32_16x16x32_bf16(
                false, al, false, Bh[s], (short)0, acc, false, false);
            acc = __builtin_amdgcn_wmma_f32_16x16x32_bf16(
                false, ah, false, Bl[s], (short)0, acc, false, false);
        }

        // Stream-ahead hint for the next tile's new keys.
        if (t0 + 640 <= T_LEN)
            __builtin_prefetch((const void*)(xb + t0 + 384 + lane * 8), 0, 1);

        // ---- store D tile: lane holds N = m, rows M = v + 8*half ----
        // Reference semantics: out[t+127] for t <= T-128; wrap tail to out[t-T].
        if (t0 + 255 <= T_LEN - 128) {
            float* op = out + xbase + t0 + 127 + m;
#pragma unroll
            for (int v = 0; v < 8; ++v)
                op[16 * (v + 8 * half)] = acc[v];
        } else {
#pragma unroll
            for (int v = 0; v < 8; ++v) {
                const long t = t0 + 16 * (v + 8 * half) + m;
                if (t <= T_LEN - 128)
                    out[xbase + t + 127] = acc[v];
                else if (t >= T_LEN && t - T_LEN <= 126)
                    out[xbase + t - T_LEN] = acc[v];
            }
        }
    }
}

// ---------------------------------------------------------------------------
extern "C" void kernel_launch(void* const* d_in, const int* in_sizes, int n_in,
                              void* d_out, int out_size, void* d_ws, size_t ws_size,
                              hipStream_t stream) {
    (void)in_sizes; (void)n_in; (void)out_size; (void)ws_size;

    const float* x   = (const float*)d_in[0];   // azimuth_data (32,1,T)
    const float* wr  = (const float*)d_in[1];   // w_real   (1,32)
    const float* wi  = (const float*)d_in[2];   // w_imag   (1,32)
    const float* ldt = (const float*)d_in[3];   // log_dt   (1,1)
    const float* Cr  = (const float*)d_in[4];   // C_real   (1,1,32)
    const float* Ci  = (const float*)d_in[5];   // C_imag   (1,1,32)
    const float* Br  = (const float*)d_in[6];   // B_real   (1,32)
    const float* Bi  = (const float*)d_in[7];   // B_imag   (1,32)
    const float* Dp  = (const float*)d_in[8];   // D        (1,)
    float* out = (float*)d_out;                 // (1,32,1,T) -> flat b*T + t

    char*   ws   = (char*)d_ws;
    float*  h_ws = (float*)ws;
    __bf16* bhi  = (__bf16*)(ws + 512);
    __bf16* blo  = (__bf16*)(ws + 512 + NCHUNK * 512 * 2);

    ssm_build_h<<<1, 256, 0, stream>>>(wr, wi, ldt, Cr, Ci, Br, Bi, Dp,
                                       h_ws, bhi, blo);

    const int nblocks = BATCH_ * SEGS_PER_BATCH / WAVES_PER_BLOCK;  // 1024
    ssm_conv_wmma<<<nblocks, 256, 0, stream>>>(x, bhi, blo, out);
}
